// SparseMoELanguageModel_58463094833558
// MI455X (gfx1250) — matrix operand order repaired and checked by reading, not compiled, not run on hardware
//
#include <hip/hip_runtime.h>

typedef __bf16 bf16;
typedef __attribute__((ext_vector_type(16))) __bf16 v16bf;
typedef __attribute__((ext_vector_type(8)))  __bf16 v8bf;
typedef __attribute__((ext_vector_type(8)))  float  v8f;

#define Bn   4
#define Tn   1024
#define Dn   1024
#define Hn   16
#define HDn  64
#define En   8
#define CAPn 1024
#define NTOK 4096
#define D3   3072
#define D4   4096
#define Ln   2

// ---------------- WMMA tile loaders (16-bit A/B, this code's contiguous k = 16*half + s in both) ----

__device__ __forceinline__ v16bf load_a16(const bf16* p) {
  // A 16x32: lane half kb = (lane>>4)*16 ; elements 0-15 = K kb..kb+15 contiguous
  v8bf lo = *(const v8bf*)(p);
  v8bf hi = *(const v8bf*)(p + 8);
  return __builtin_shufflevector(lo, hi, 0,1,2,3,4,5,6,7,8,9,10,11,12,13,14,15);
}
__device__ __forceinline__ v16bf load_b16(const bf16* p) {
  // B 32x16: lane = column; lane half holds K = half*16 .. half*16+15 contiguous
  return *(const v16bf*)(p);
}

// ---------------- Generic WMMA GEMM:  C[z] = A[z] (MxK, bf16) * Bt[z]^T (Bt is NxK) --
// Wave tile: 64x64 (4x4 WMMA), block tile: 128x256 (8 waves as 2x4).
// LDA / LDBT / KD are compile-time so tile base addresses are loop-invariant and
// k-advance folds into immediate offsets (fewer VALU addr ops + hazard nops).

enum { EPI_F32 = 0, EPI_BF16 = 1, EPI_RESID = 2, EPI_GELU = 3, EPI_MOE = 4 };

template<int EPI, int LDA, int LDBT, int KD>
__global__ __launch_bounds__(256) void wmma_gemm(
    const bf16* __restrict__ A,  long long sA,
    const bf16* __restrict__ Bt, long long sB,
    void* __restrict__ C,        long long sC, int ldc,
    int M, int N, int causal,
    const float* __restrict__ bias, long long sBias,
    const int* __restrict__ tok_map, const float* __restrict__ gate_val, long long sRoute,
    float* __restrict__ aout, int aldc)
{
  if (causal && (int)blockIdx.x * 256 > (int)blockIdx.y * 128 + 127) return;
  const int z = blockIdx.z;
  A  += (long long)z * sA;
  Bt += (long long)z * sB;
  const int lane = threadIdx.x & 31;
  const int wid  = threadIdx.x >> 5;
  const int m0 = blockIdx.y * 128 + (wid >> 2) * 64;
  const int n0 = blockIdx.x * 256 + (wid & 3) * 64;
  if (m0 >= M || n0 >= N) return;

  const int lm   = lane & 15;
  const int half = lane >> 4;
  const int kbA  = half * 16;
  const int kbB  = half * 16;

  // loop-invariant per-tile base pointers
  const bf16* ap[4];
  const bf16* bp[4];
#pragma unroll
  for (int i = 0; i < 4; ++i) ap[i] = A + (size_t)(m0 + i * 16 + lm) * LDA + kbA;
#pragma unroll
  for (int j = 0; j < 4; ++j) bp[j] = Bt + (size_t)(n0 + j * 16 + lm) * LDBT + kbB;

  v8f acc[4][4] = {};
  for (int kk = 0; kk < KD; kk += 32) {
    v16bf a[4], b[4];
#pragma unroll
    for (int i = 0; i < 4; ++i) a[i] = load_a16(ap[i] + kk);
#pragma unroll
    for (int j = 0; j < 4; ++j) b[j] = load_b16(bp[j] + kk);
#pragma unroll
    for (int i = 0; i < 4; ++i)
#pragma unroll
      for (int j = 0; j < 4; ++j)
        acc[i][j] = __builtin_amdgcn_wmma_f32_16x16x32_bf16(
            false, a[i], false, b[j], (short)0, acc[i][j], false, false);
  }

  // C/D layout: col = lm, row = half*8 + r
#pragma unroll
  for (int ti = 0; ti < 4; ++ti)
#pragma unroll
    for (int tj = 0; tj < 4; ++tj) {
      const int colg = n0 + tj * 16 + lm;
#pragma unroll
      for (int r = 0; r < 8; ++r) {
        const int rowg = m0 + ti * 16 + half * 8 + r;
        float v = acc[ti][tj][r];
        if constexpr (EPI == EPI_F32) {
          ((float*)C)[(long long)z * sC + (size_t)rowg * ldc + colg] = v;
        } else if constexpr (EPI == EPI_BF16) {
          ((bf16*)C)[(long long)z * sC + (size_t)rowg * ldc + colg] = (bf16)v;
        } else if constexpr (EPI == EPI_RESID) {
          float* cp = (float*)C + (long long)z * sC + (size_t)rowg * ldc + colg;
          *cp = v + *cp;
        } else if constexpr (EPI == EPI_GELU) {
          float t = v + bias[(long long)z * sBias + colg];
          float g = 0.5f * t * (1.0f + erff(t * 0.70710678118654752f));
          ((bf16*)C)[(long long)z * sC + (size_t)rowg * ldc + colg] = (bf16)g;
        } else { // EPI_MOE
          int tok = tok_map[(long long)z * sRoute + rowg];
          if (tok >= 0) {
            float gv = gate_val[(long long)z * sRoute + rowg];
            float t = (v + bias[(long long)z * sBias + colg]) * gv;
            atomicAdd(aout + (size_t)tok * aldc + colg, t);
          }
        }
      }
    }
}

// ---------------- Support kernels ------------------------------------------------

__global__ void k_transpose_bf16(const float* __restrict__ w, bf16* __restrict__ wt,
                                 int R, int Cc, long long total) {
  long long i = (long long)blockIdx.x * blockDim.x + threadIdx.x;
  long long stride = (long long)gridDim.x * blockDim.x;
  long long rc = (long long)R * Cc;
  for (; i < total; i += stride) {
    long long b = i / rc; long long rem = i - b * rc;
    int r = (int)(rem / Cc), c = (int)(rem - (long long)r * Cc);
    wt[b * rc + (long long)c * R + r] = (bf16)w[i];
  }
}

__global__ __launch_bounds__(256) void k_layernorm(const float* __restrict__ x,
    const float* __restrict__ g, const float* __restrict__ b, bf16* __restrict__ out) {
  __shared__ float s1[256], s2[256];
  const int row = blockIdx.x, tid = threadIdx.x;
  const float* xr = x + (size_t)row * Dn;
  float a1 = 0.f, a2 = 0.f;
  for (int d = tid; d < Dn; d += 256) { float v = xr[d]; a1 += v; a2 += v * v; }
  s1[tid] = a1; s2[tid] = a2; __syncthreads();
  for (int s = 128; s > 0; s >>= 1) {
    if (tid < s) { s1[tid] += s1[tid + s]; s2[tid] += s2[tid + s]; }
    __syncthreads();
  }
  const float mu = s1[0] * (1.0f / Dn);
  const float var = s2[0] * (1.0f / Dn) - mu * mu;
  const float rstd = rsqrtf(var + 1e-5f);
  for (int d = tid; d < Dn; d += 256)
    out[(size_t)row * Dn + d] = (bf16)((xr[d] - mu) * rstd * g[d] + b[d]);
}

__global__ __launch_bounds__(256) void k_rope(const float* __restrict__ qkv,
    bf16* __restrict__ Q, bf16* __restrict__ K, bf16* __restrict__ VT) {
  const int idx = blockIdx.x * 256 + threadIdx.x;   // B*T*H*32 = 2^21
  const int i = idx & 31;
  const int h = (idx >> 5) & 15;
  const int t = (idx >> 9) & 1023;
  const int b = idx >> 19;
  const size_t base = ((size_t)(b * Tn + t)) * D3 + (size_t)h * (3 * HDn);
  const float ang = (float)t * powf(10000.0f, -(float)i * (1.0f / 32.0f));
  const float s = sinf(ang), c = cosf(ang);
  const float q1 = qkv[base + i],       q2 = qkv[base + 32 + i];
  const float k1 = qkv[base + 64 + i],  k2 = qkv[base + 96 + i];
  const float v1 = qkv[base + 128 + i], v2 = qkv[base + 160 + i];
  const size_t qb = ((size_t)((b * Hn + h) * Tn + t)) * HDn;
  Q[qb + i]      = (bf16)(q1 * c - q2 * s);
  Q[qb + 32 + i] = (bf16)(q1 * s + q2 * c);
  K[qb + i]      = (bf16)(k1 * c - k2 * s);
  K[qb + 32 + i] = (bf16)(k1 * s + k2 * c);
  const size_t vtb = ((size_t)(b * Hn + h)) * HDn * Tn;
  VT[vtb + (size_t)i * Tn + t]        = (bf16)v1;
  VT[vtb + (size_t)(i + 32) * Tn + t] = (bf16)v2;
}

__global__ __launch_bounds__(256) void k_softmax_causal(const float* __restrict__ S,
                                                        bf16* __restrict__ P) {
  const int q = blockIdx.x, h = blockIdx.y, tid = threadIdx.x;
  const float* sr = S + ((size_t)h * Tn + q) * Tn;
  bf16* pr = P + ((size_t)h * Tn + q) * Tn;
  __shared__ float red[256];
  float m = -3.0e38f;
  for (int k = tid; k <= q; k += 256) m = fmaxf(m, sr[k] * 0.125f);
  red[tid] = m; __syncthreads();
  for (int s = 128; s > 0; s >>= 1) { if (tid < s) red[tid] = fmaxf(red[tid], red[tid + s]); __syncthreads(); }
  m = red[0]; __syncthreads();
  float sum = 0.f;
  for (int k = tid; k <= q; k += 256) sum += expf(sr[k] * 0.125f - m);
  red[tid] = sum; __syncthreads();
  for (int s = 128; s > 0; s >>= 1) { if (tid < s) red[tid] += red[tid + s]; __syncthreads(); }
  const float inv = 1.0f / red[0];
  for (int k = tid; k < Tn; k += 256)
    pr[k] = (bf16)((k <= q) ? expf(sr[k] * 0.125f - m) * inv : 0.0f);
}

__global__ __launch_bounds__(256) void k_gate(const bf16* __restrict__ xb,
    const float* __restrict__ gw, float* __restrict__ logits) {
  __shared__ float sh[256];
  const int tok = blockIdx.x, tid = threadIdx.x;
  const int e = tid & 7, dg = tid >> 3;
  const bf16* xr = xb + (size_t)tok * Dn;
  float acc = 0.f;
  for (int d = dg; d < Dn; d += 32) acc += (float)xr[d] * gw[(size_t)d * En + e];
  sh[tid] = acc; __syncthreads();
  for (int s = 128; s >= 8; s >>= 1) { if (tid < s) sh[tid] += sh[tid + s]; __syncthreads(); }
  if (tid < 8) logits[(size_t)tok * En + tid] = sh[tid];
}

__global__ void k_topk(const float* __restrict__ logits, float* __restrict__ gates,
                       int* __restrict__ inds) {
  const int tok = blockIdx.x * blockDim.x + threadIdx.x;
  if (tok >= NTOK) return;
  float l[En]; float m = -3.0e38f;
  for (int e = 0; e < En; ++e) { l[e] = logits[tok * En + e]; m = fmaxf(m, l[e]); }
  float s = 0.f;
  for (int e = 0; e < En; ++e) { l[e] = expf(l[e] - m); s += l[e]; }
  const float inv = 1.0f / s;
  int i0 = 0; float t0 = -1.f;
  for (int e = 0; e < En; ++e) { l[e] *= inv; gates[tok * En + e] = l[e]; if (l[e] > t0) { t0 = l[e]; i0 = e; } }
  int i1 = 0; float t1 = -1.f;
  for (int e = 0; e < En; ++e) { if (e != i0 && l[e] > t1) { t1 = l[e]; i1 = e; } }
  inds[tok * 2] = i0; inds[tok * 2 + 1] = i1;
}

__global__ __launch_bounds__(256) void k_route(const int* __restrict__ inds,
    const float* __restrict__ gates, int* __restrict__ map, float* __restrict__ gv) {
  const int e = blockIdx.x, tid = threadIdx.x;
  for (int s = tid; s < CAPn; s += 256) { map[e * CAPn + s] = -1; gv[e * CAPn + s] = 0.f; }
  __syncthreads();
  if (tid == 0) {
    int cnt = 0;
    for (int tok = 0; tok < NTOK && cnt < CAPn; ++tok) {
      if (inds[tok * 2] == e || inds[tok * 2 + 1] == e) {
        map[e * CAPn + cnt] = tok;
        gv[e * CAPn + cnt] = gates[tok * En + e];
        ++cnt;
      }
    }
  }
}

__global__ __launch_bounds__(256) void k_gather(const bf16* __restrict__ xb,
    const int* __restrict__ map, bf16* __restrict__ xg) {
  const int slot = blockIdx.x;
  const int tok = map[slot];
  bf16* dst = xg + (size_t)slot * Dn;
  if (tok >= 0) {
    const bf16* src = xb + (size_t)tok * Dn;
    for (int d = threadIdx.x; d < Dn; d += 256) dst[d] = src[d];
  } else {
    for (int d = threadIdx.x; d < Dn; d += 256) dst[d] = (bf16)0.0f;
  }
}

__global__ void k_zero(float* p, long long n) {
  long long i = (long long)blockIdx.x * blockDim.x + threadIdx.x;
  const long long st = (long long)gridDim.x * blockDim.x;
  for (; i < n; i += st) p[i] = 0.f;
}
__global__ void k_add(float* __restrict__ x, const float* __restrict__ y, long long n) {
  long long i = (long long)blockIdx.x * blockDim.x + threadIdx.x;
  const long long st = (long long)gridDim.x * blockDim.x;
  for (; i < n; i += st) x[i] += y[i];
}

// ---------------- Host side ------------------------------------------------------

static constexpr size_t AL(size_t x) { return (x + 255) & ~(size_t)255; }

template<int EPI, int LDA, int LDBT, int KD>
static void launch_gemm(hipStream_t st,
    const bf16* A, long long sA,
    const bf16* Bt, long long sB,
    void* C, long long sC, int ldc,
    int M, int N, int nz, int causal,
    const float* bias = nullptr, long long sBias = 0,
    const int* map = nullptr, const float* gv = nullptr, long long sRoute = 0,
    float* aout = nullptr, int aldc = 0)
{
  dim3 grid((N + 255) / 256, (M + 127) / 128, nz);
  wmma_gemm<EPI, LDA, LDBT, KD><<<grid, 256, 0, st>>>(A, sA, Bt, sB, C, sC, ldc,
                                                      M, N, causal, bias, sBias,
                                                      map, gv, sRoute, aout, aldc);
}

extern "C" void kernel_launch(void* const* d_in, const int* in_sizes, int n_in,
                              void* d_out, int out_size, void* d_ws, size_t ws_size,
                              hipStream_t stream) {
  const float* x_in   = (const float*)d_in[0];
  const float* ln1_g  = (const float*)d_in[1];
  const float* ln1_b  = (const float*)d_in[2];
  const float* qkv_w  = (const float*)d_in[3];
  const float* proj_w = (const float*)d_in[4];
  const float* ln2_g  = (const float*)d_in[5];
  const float* ln2_b  = (const float*)d_in[6];
  const float* gate_w = (const float*)d_in[7];
  const float* w1     = (const float*)d_in[8];
  const float* b1     = (const float*)d_in[9];
  const float* w2     = (const float*)d_in[10];
  const float* b2     = (const float*)d_in[11];
  float* x = (float*)d_out;

  // ---- workspace carving ----
  char* p = (char*)d_ws;
  auto take = [&](size_t s) { char* r = p; p += AL(s); return r; };
  bf16* WQKVT  = (bf16*)take((size_t)D3 * Dn * 2);     // [3072 x 1024]
  bf16* WPROJT = (bf16*)take((size_t)Dn * Dn * 2);     // [1024 x 1024]
  bf16* WW1T   = (bf16*)take((size_t)En * D4 * Dn * 2);// [E][4096 x 1024]
  bf16* WW2T   = (bf16*)take((size_t)En * Dn * D4 * 2);// [E][1024 x 4096]
  bf16* LNBF   = (bf16*)take((size_t)NTOK * Dn * 2);
  char* scratch = p;
  // attention phase view
  char* q = scratch;
  auto takeA = [&](size_t s) { char* r = q; q += AL(s); return r; };
  float* QKVF  = (float*)takeA((size_t)NTOK * D3 * 4);
  bf16*  Qb    = (bf16*)takeA((size_t)Bn * Hn * Tn * HDn * 2);
  bf16*  Kb    = (bf16*)takeA((size_t)Bn * Hn * Tn * HDn * 2);
  bf16*  VTb   = (bf16*)takeA((size_t)Bn * Hn * HDn * Tn * 2);
  float* SCORE = (float*)takeA((size_t)Hn * Tn * Tn * 4);
  bf16*  PROBS = (bf16*)takeA((size_t)Hn * Tn * Tn * 2);
  bf16*  ATTNB = (bf16*)takeA((size_t)NTOK * Dn * 2);
  // moe phase view (reuses scratch)
  q = scratch;
  float* LOGIT = (float*)takeA((size_t)NTOK * En * 4);
  float* GATES = (float*)takeA((size_t)NTOK * En * 4);
  int*   INDS  = (int*)takeA((size_t)NTOK * 2 * 4);
  int*   TOKM  = (int*)takeA((size_t)En * CAPn * 4);
  float* GVAL  = (float*)takeA((size_t)En * CAPn * 4);
  bf16*  XG    = (bf16*)takeA((size_t)En * CAPn * Dn * 2);
  bf16*  HBUF  = (bf16*)takeA((size_t)En * CAPn * D4 * 2);
  float* MOEO  = (float*)takeA((size_t)NTOK * Dn * 4);

  hipMemcpyAsync(x, x_in, (size_t)NTOK * Dn * 4, hipMemcpyDeviceToDevice, stream);

  for (int l = 0; l < Ln; ++l) {
    // ---- per-layer weight conversion (f32 -> bf16, transposed to [N x K]) ----
    k_transpose_bf16<<<8192, 256, 0, stream>>>(qkv_w + (size_t)l * Dn * D3, WQKVT, Dn, D3, (long long)Dn * D3);
    k_transpose_bf16<<<8192, 256, 0, stream>>>(proj_w + (size_t)l * Dn * Dn, WPROJT, Dn, Dn, (long long)Dn * Dn);
    k_transpose_bf16<<<8192, 256, 0, stream>>>(w1 + (size_t)l * En * Dn * D4, WW1T, Dn, D4, (long long)En * Dn * D4);
    k_transpose_bf16<<<8192, 256, 0, stream>>>(w2 + (size_t)l * En * D4 * Dn, WW2T, D4, Dn, (long long)En * D4 * Dn);

    // ---- attention ----
    k_layernorm<<<NTOK, 256, 0, stream>>>(x, ln1_g + l * Dn, ln1_b + l * Dn, LNBF);
    launch_gemm<EPI_F32, Dn, Dn, Dn>(stream, LNBF, 0, WQKVT, 0, QKVF, 0, D3,
                                     NTOK, D3, 1, 0);
    k_rope<<<(Bn * Tn * Hn * 32) / 256, 256, 0, stream>>>(QKVF, Qb, Kb, VTb);

    for (int b = 0; b < Bn; ++b) {
      const long long hs = (long long)Tn * HDn;
      launch_gemm<EPI_F32, HDn, HDn, HDn>(stream,
          Qb + (size_t)b * Hn * Tn * HDn, hs,
          Kb + (size_t)b * Hn * Tn * HDn, hs,
          SCORE, (long long)Tn * Tn, Tn,
          Tn, Tn, Hn, /*causal=*/1);
      k_softmax_causal<<<dim3(Tn, Hn), 256, 0, stream>>>(SCORE, PROBS);
      launch_gemm<EPI_BF16, Tn, Tn, Tn>(stream,
          PROBS, (long long)Tn * Tn,
          VTb + (size_t)b * Hn * HDn * Tn, (long long)HDn * Tn,
          ATTNB + (size_t)b * Tn * Dn, (long long)HDn, Dn,
          Tn, HDn, Hn, 0);
    }
    launch_gemm<EPI_RESID, Dn, Dn, Dn>(stream, ATTNB, 0, WPROJT, 0, x, 0, Dn,
                                       NTOK, Dn, 1, 0);

    // ---- MoE ----
    k_layernorm<<<NTOK, 256, 0, stream>>>(x, ln2_g + l * Dn, ln2_b + l * Dn, LNBF);
    k_gate<<<NTOK, 256, 0, stream>>>(LNBF, gate_w + (size_t)l * Dn * En, LOGIT);
    k_topk<<<NTOK / 256, 256, 0, stream>>>(LOGIT, GATES, INDS);
    k_route<<<En, 256, 0, stream>>>(INDS, GATES, TOKM, GVAL);
    k_gather<<<En * CAPn, 256, 0, stream>>>(LNBF, TOKM, XG);
    k_zero<<<4096, 256, 0, stream>>>(MOEO, (long long)NTOK * Dn);

    launch_gemm<EPI_GELU, Dn, Dn, Dn>(stream,
        XG, (long long)CAPn * Dn,
        WW1T, (long long)D4 * Dn,
        HBUF, (long long)CAPn * D4, D4,
        CAPn, D4, En, 0,
        b1 + (size_t)l * En * D4, D4);

    launch_gemm<EPI_MOE, D4, D4, D4>(stream,
        HBUF, (long long)CAPn * D4,
        WW2T, (long long)Dn * D4,
        nullptr, 0, 0,
        CAPn, Dn, En, 0,
        b2 + (size_t)l * En * Dn, Dn,
        TOKM, GVAL, CAPn,
        MOEO, Dn);

    k_add<<<4096, 256, 0, stream>>>(x, MOEO, (long long)NTOK * Dn);
  }
}
